// MoE_73753178407160
// MI455X (gfx1250) — compile-verified, matrix operand order checked by reading, not run on hardware
//
#include <hip/hip_runtime.h>
#include <hip/hip_bf16.h>

typedef __attribute__((ext_vector_type(16))) __bf16 v16bf;
typedef __attribute__((ext_vector_type(8)))  float  v8f;
typedef __attribute__((ext_vector_type(4)))  unsigned tdm_g0_t;
typedef __attribute__((ext_vector_type(8)))  int      tdm_g1_t;
typedef __attribute__((ext_vector_type(4)))  int      tdm_g2_t;
typedef __attribute__((ext_vector_type(8)))  int      tdm_g4_t;

namespace moecfg {
constexpr int Nn  = 8192;   // B*S tokens
constexpr int Hh  = 1024;   // hidden
constexpr int Ee  = 8;      // routed experts
constexpr int Kk  = 2;      // top-k
constexpr int Ii  = 1024;   // routed intermediate
constexpr int ISh = 2048;   // shared intermediate (NSH*I)
constexpr int CAP = 2560;   // ceil(1.25 * Nn * Kk / Ee), multiple of 32
constexpr float RSF = 1.0f;
}
using namespace moecfg;

// ---------------------------------------------------------------------------
// CDNA5 async global->LDS 16B copy (GLOBAL_LOAD_ASYNC_TO_LDS_B128, ASYNCcnt).
// ---------------------------------------------------------------------------
static __device__ inline void async_cp16(const __bf16* g, __bf16* lds) {
    const unsigned loff = (unsigned)(uintptr_t)lds;
    asm volatile("global_load_async_to_lds_b128 %0, %1, off"
                 :: "v"(loff), "v"(g) : "memory");
}

static __device__ inline void wait_async0() {
#if __has_builtin(__builtin_amdgcn_s_wait_asynccnt)
    __builtin_amdgcn_s_wait_asynccnt(0);
#else
    asm volatile("s_wait_asynccnt 0x0" ::: "memory");
#endif
}

static __device__ inline void wait_tensor0() {
#if __has_builtin(__builtin_amdgcn_s_wait_tensorcnt)
    __builtin_amdgcn_s_wait_tensorcnt(0);
#else
    asm volatile("s_wait_tensorcnt 0x0" ::: "memory");
#endif
}

// ---------------------------------------------------------------------------
// TDM 2D tile load: rows x 32 bf16 tile -> LDS with 80B row stride.
// D# group0: count=1, lds_addr, global_addr(57b), type=2.
// D# group1: data_size=1(2B), pad_enable, pad_interval=3 (16 DW = 64B row),
//            pad_amount=3 (4 DW = 16B pad) -> LDS row stride 40 elements.
// Toolchain here declares the 6-arg builtin (g0, g1, g2, g3, g4, cpol).
// ---------------------------------------------------------------------------
static __device__ inline void tdm_load_tile32(
    const __bf16* gbase, __bf16* lds, int rows_rem, int k_rem, int kdim,
    int tile_rows)
{
    const unsigned long long ga = (unsigned long long)(uintptr_t)gbase;
    tdm_g0_t g0;
    g0[0] = 1u;                                             // count=1, no gather
    g0[1] = (unsigned)(uintptr_t)lds;                       // lds_addr (bytes)
    g0[2] = (unsigned)(ga & 0xffffffffu);                   // global_addr lo
    g0[3] = (unsigned)((ga >> 32) & 0x01ffffffu) | 0x80000000u;  // hi | type=2
    tdm_g1_t g1;
    g1[0] = (int)((1u << 16) | (1u << 20) | (3u << 22) | (3u << 25));
    g1[1] = (int)(((unsigned)k_rem & 0xffffu) << 16);       // tensor_dim0 lo16
    g1[2] = (int)((((unsigned)k_rem >> 16) & 0xffffu) |
                  (((unsigned)rows_rem & 0xffffu) << 16));  // dim0 hi / dim1 lo
    g1[3] = (int)((((unsigned)rows_rem >> 16) & 0xffffu) |
                  (32u << 16));                             // dim1 hi / tile_dim0=32
    g1[4] = tile_rows;                                      // tile_dim1, tile_dim2=0
    g1[5] = kdim;                                           // dim0_stride lo32
    g1[6] = 0;                                              // dim0_stride hi / dim1_stride lo
    g1[7] = 0;                                              // dim1_stride hi
#if __has_builtin(__builtin_amdgcn_tensor_load_to_lds)
    const tdm_g2_t z4 = {0, 0, 0, 0};
    const tdm_g4_t z8 = {0, 0, 0, 0, 0, 0, 0, 0};
    __builtin_amdgcn_tensor_load_to_lds(g0, g1, z4, z4, z8, 0);
#else
    asm volatile("tensor_load_to_lds %0, %1" :: "s"(g0), "s"(g1) : "memory");
#endif
}

// ---------------------------------------------------------------------------
// 16x32 bf16 fragment row (A-layout, ISA 7.12.2).
// ---------------------------------------------------------------------------
static __device__ inline v16bf frag16(const __bf16* rowp, int kh) {
    union { uint4 u[2]; v16bf v; } t;
    t.u[0] = *(const uint4*)(rowp + kh);
    t.u[1] = *(const uint4*)(rowp + 16 + kh);
    return t.v;
}

// ---------------------------------------------------------------------------
// Kernel 0: one-time fp32 -> bf16 conversion (weights + activations).
// ---------------------------------------------------------------------------
__global__ __launch_bounds__(256) void moe_cvt_bf16(
    const float* __restrict__ s, __bf16* __restrict__ d, int n)
{
    const int i = (blockIdx.x * 256 + threadIdx.x) * 4;
    if (i < n) {
        const float4 v = *(const float4*)(s + i);
        d[i + 0] = (__bf16)v.x; d[i + 1] = (__bf16)v.y;
        d[i + 2] = (__bf16)v.z; d[i + 3] = (__bf16)v.w;
    }
}

// ---------------------------------------------------------------------------
// Kernel 1: router. One wave per token.
// ---------------------------------------------------------------------------
__global__ __launch_bounds__(256) void moe_router(
    const float* __restrict__ x, const float* __restrict__ gw,
    int* __restrict__ tk_idx, float* __restrict__ tk_w)
{
    const int lane = threadIdx.x & 31;
    const int wv   = threadIdx.x >> 5;
    const int tok  = blockIdx.x * 8 + wv;
    const float* xr = x + (size_t)tok * Hh;

    float xv[32];
#pragma unroll
    for (int j = 0; j < 32; ++j) xv[j] = xr[lane + 32 * j];

    float lg[Ee];
#pragma unroll
    for (int e = 0; e < Ee; ++e) {
        const float* gr = gw + (size_t)e * Hh;
        float acc = 0.f;
#pragma unroll
        for (int j = 0; j < 32; ++j) acc += xv[j] * gr[lane + 32 * j];
#pragma unroll
        for (int off = 16; off >= 1; off >>= 1)
            acc += __shfl_xor(acc, off, 32);
        lg[e] = acc;
    }

    float mx = lg[0];
#pragma unroll
    for (int e = 1; e < Ee; ++e) mx = fmaxf(mx, lg[e]);
    float sc[Ee]; float sum = 0.f;
#pragma unroll
    for (int e = 0; e < Ee; ++e) { sc[e] = __expf(lg[e] - mx); sum += sc[e]; }
    const float inv = 1.f / sum;
#pragma unroll
    for (int e = 0; e < Ee; ++e) sc[e] *= inv;

    int i1 = 0; float s1 = sc[0];
#pragma unroll
    for (int e = 1; e < Ee; ++e) if (sc[e] > s1) { s1 = sc[e]; i1 = e; }
    int i2 = -1; float s2 = -1.f;
#pragma unroll
    for (int e = 0; e < Ee; ++e) if (e != i1 && sc[e] > s2) { s2 = sc[e]; i2 = e; }

    const float dn = RSF / (s1 + s2 + 1e-20f);
    if (lane == 0) {
        tk_idx[tok * 2 + 0] = i1;
        tk_idx[tok * 2 + 1] = i2;
        tk_w [tok * 2 + 0] = s1 * dn;
        tk_w [tok * 2 + 1] = s2 * dn;
    }
}

// ---------------------------------------------------------------------------
// Kernel 2: ordered capacity scan + per-expert gather lists (wave32 ballot).
// ---------------------------------------------------------------------------
__global__ __launch_bounds__(256) void moe_scan(
    const int* __restrict__ tk_idx, const float* __restrict__ tk_w,
    int* __restrict__ etok, float* __restrict__ ewt, int* __restrict__ ecnt)
{
    const int lane = threadIdx.x & 31;
    const int e    = threadIdx.x >> 5;
    int cnt = 0;
    const unsigned below = (lane == 0) ? 0u : (0xffffffffu >> (32 - lane));
    for (int base = 0; base < Nn * Kk; base += 32) {
        const int slot = base + lane;
        const bool match = (tk_idx[slot] == e);
        const unsigned m = __builtin_amdgcn_ballot_w32(match);
        const int pos = cnt + __popc(m & below);
        if (match && pos < CAP) {
            etok[e * CAP + pos] = slot >> 1;
            ewt [e * CAP + pos] = tk_w[slot];
        }
        cnt += __popc(m);
    }
    if (lane == 0) ecnt[e] = cnt < CAP ? cnt : CAP;
}

// ---------------------------------------------------------------------------
// Kernel 3: SwiGLU up-projection. A rows via per-lane async gather copies,
// B tiles (gate & up, 128x32) via TDM descriptors issued by wave 0.
// Double-buffered LDS, copies overlapped with WMMA.
// ---------------------------------------------------------------------------
template<bool GATHER>
__global__ __launch_bounds__(256) void moe_gemm_swiglu(
    const __bf16* __restrict__ xb,
    const __bf16* __restrict__ wg_all, const __bf16* __restrict__ wu_all,
    __bf16* __restrict__ out_all,
    const int* __restrict__ tok_all, const int* __restrict__ cnt_all,
    int Ntot, int Kdim, int Mrows)
{
    __shared__ __align__(16) __bf16 sA [2][32][40];
    __shared__ __align__(16) __bf16 sBg[2][128][40];
    __shared__ __align__(16) __bf16 sBu[2][128][40];

    const int e = GATHER ? blockIdx.z : 0;
    const __bf16* wg = wg_all + (size_t)e * Ntot * Kdim;
    const __bf16* wu = wu_all + (size_t)e * Ntot * Kdim;
    __bf16* out = out_all + (size_t)e * CAP * Ntot;
    const int* tl = GATHER ? (tok_all + e * CAP) : nullptr;

    int cnt = Mrows;
    if (GATHER) { const int c = cnt_all[e]; cnt = c < CAP ? c : CAP; }
    const int m_base = blockIdx.y * 32;
    const int cnt_round = (cnt + 31) & ~31;
    if (m_base >= cnt_round) return;
    const int n_base = blockIdx.x * 128;

    const int lane = threadIdx.x & 31;
    const int wv   = threadIdx.x >> 5;
    const int wm   = wv >> 2, wn = wv & 3;

    auto issue = [&](int k0, int buf) {
        if (threadIdx.x < 128) {              // A: 32x32 bf16 = 128 x 16B, gathered
            const int row = threadIdx.x >> 2, seg = (threadIdx.x & 3) * 8;
            const int p = m_base + row;
            int tok = p;
            if (GATHER) tok = (p < cnt) ? tl[p] : 0;
            async_cp16(xb + (size_t)tok * Kdim + k0 + seg, &sA[buf][row][seg]);
        }
        if (wv == 0) {                        // B tiles: TDM DMA, one wave issues
            const size_t goff = (size_t)n_base * Kdim + k0;
            tdm_load_tile32(wg + goff, &sBg[buf][0][0],
                            Ntot - n_base, Kdim - k0, Kdim, 128);
            tdm_load_tile32(wu + goff, &sBu[buf][0][0],
                            Ntot - n_base, Kdim - k0, Kdim, 128);
        }
    };

    v8f ag[2] = {}, au[2] = {};
    const int T = Kdim / 32;
    issue(0, 0);

    for (int it = 0; it < T; ++it) {
        const int cur = it & 1;
        wait_async0();                         // own wave's A copies
        wait_tensor0();                        // wave 0's TDM tiles
        __syncthreads();                       // all waves ready, prev reads done
        if (it + 1 < T) issue((it + 1) * 32, cur ^ 1);   // overlap next copies

        const int kh = (lane >> 4) * 8;
        const v16bf a = frag16(&sA[cur][wm * 16 + (lane & 15)][0], kh);
#pragma unroll
        for (int sub = 0; sub < 2; ++sub) {
            const int brow = wn * 32 + sub * 16 + (lane & 15);
            const v16bf bg = frag16(&sBg[cur][brow][0], kh);
            const v16bf bu = frag16(&sBu[cur][brow][0], kh);
            ag[sub] = __builtin_amdgcn_wmma_f32_16x16x32_bf16(
                false, a, false, bg, (short)0, ag[sub], false, false);
            au[sub] = __builtin_amdgcn_wmma_f32_16x16x32_bf16(
                false, a, false, bu, (short)0, au[sub], false, false);
        }
    }

#pragma unroll
    for (int sub = 0; sub < 2; ++sub) {
        const int ncol = n_base + wn * 32 + sub * 16 + (lane & 15);
#pragma unroll
        for (int i = 0; i < 8; ++i) {          // C layout: VGPR i -> M=i / i+8
            const int mrow = m_base + wm * 16 + i + ((lane >> 4) << 3);
            const float g = ag[sub][i];
            const float u = au[sub][i];
            const float s = g * __builtin_amdgcn_rcpf(1.f + __expf(-g)) * u;
            out[(size_t)mrow * Ntot + ncol] = (__bf16)s;
        }
    }
}

// ---------------------------------------------------------------------------
// Kernel 4: down-projection. A via per-lane async copies, B via TDM.
// SCATTER=false: plain store into y. SCATTER=true: weighted atomic add.
// ---------------------------------------------------------------------------
template<bool SCATTER>
__global__ __launch_bounds__(256) void moe_gemm_down(
    const __bf16* __restrict__ h_all,
    const __bf16* __restrict__ wd_all,
    float* __restrict__ y,
    const int* __restrict__ tok_all, const float* __restrict__ wt_all,
    const int* __restrict__ cnt_all,
    int Ntot, int Kdim, int Mrows)
{
    __shared__ __align__(16) __bf16 sA[2][32][40];
    __shared__ __align__(16) __bf16 sB[2][128][40];

    const int e = SCATTER ? blockIdx.z : 0;
    const __bf16* h  = h_all  + (size_t)e * CAP * Kdim;
    const __bf16* wd = wd_all + (size_t)e * Ntot * Kdim;
    const int*   tl = SCATTER ? (tok_all + e * CAP) : nullptr;
    const float* wl = SCATTER ? (wt_all + e * CAP) : nullptr;

    int cnt = Mrows;
    if (SCATTER) { const int c = cnt_all[e]; cnt = c < CAP ? c : CAP; }
    const int m_base = blockIdx.y * 32;
    const int cnt_round = (cnt + 31) & ~31;
    if (m_base >= cnt_round) return;
    const int n_base = blockIdx.x * 128;

    const int lane = threadIdx.x & 31;
    const int wv   = threadIdx.x >> 5;
    const int wm   = wv >> 2, wn = wv & 3;

    auto issue = [&](int k0, int buf) {
        if (threadIdx.x < 128) {              // A: 32x32 bf16 = 128 x 16B
            const int row = threadIdx.x >> 2, seg = (threadIdx.x & 3) * 8;
            async_cp16(h + (size_t)(m_base + row) * Kdim + k0 + seg,
                       &sA[buf][row][seg]);
        }
        if (wv == 0) {                        // B tile: TDM DMA
            tdm_load_tile32(wd + (size_t)n_base * Kdim + k0, &sB[buf][0][0],
                            Ntot - n_base, Kdim - k0, Kdim, 128);
        }
    };

    v8f acc[2] = {};
    const int T = Kdim / 32;
    issue(0, 0);

    for (int it = 0; it < T; ++it) {
        const int cur = it & 1;
        wait_async0();
        wait_tensor0();
        __syncthreads();
        if (it + 1 < T) issue((it + 1) * 32, cur ^ 1);

        const int kh = (lane >> 4) * 8;
        const v16bf a = frag16(&sA[cur][wm * 16 + (lane & 15)][0], kh);
#pragma unroll
        for (int sub = 0; sub < 2; ++sub) {
            const v16bf b = frag16(&sB[cur][wn * 32 + sub * 16 + (lane & 15)][0], kh);
            acc[sub] = __builtin_amdgcn_wmma_f32_16x16x32_bf16(
                false, a, false, b, (short)0, acc[sub], false, false);
        }
    }

#pragma unroll
    for (int sub = 0; sub < 2; ++sub) {
        const int ncol = n_base + wn * 32 + sub * 16 + (lane & 15);
#pragma unroll
        for (int i = 0; i < 8; ++i) {
            const int p = m_base + wm * 16 + i + ((lane >> 4) << 3);
            const float v = acc[sub][i];
            if (SCATTER) {
                if (p < cnt) {
                    const int tok = tl[p];
                    atomicAdd(&y[(size_t)tok * Ntot + ncol], wl[p] * v);
                }
            } else {
                y[(size_t)p * Ntot + ncol] = v;
            }
        }
    }
}

// ---------------------------------------------------------------------------
// Host: workspace carve + launch sequence (all on `stream`, capture-safe).
// ---------------------------------------------------------------------------
extern "C" void kernel_launch(void* const* d_in, const int* in_sizes, int n_in,
                              void* d_out, int out_size, void* d_ws, size_t ws_size,
                              hipStream_t stream) {
    (void)in_sizes; (void)n_in; (void)out_size; (void)ws_size;

    const float* x       = (const float*)d_in[0];   // [N, H]
    const float* gate_w  = (const float*)d_in[1];   // [E, H]
    const float* we_gate = (const float*)d_in[2];   // [E, I, H]
    const float* we_up   = (const float*)d_in[3];   // [E, I, H]
    const float* we_down = (const float*)d_in[4];   // [E, H, I]
    const float* ws_gate = (const float*)d_in[5];   // [ISh, H]
    const float* ws_up   = (const float*)d_in[6];   // [ISh, H]
    const float* ws_down = (const float*)d_in[7];   // [H, ISh]
    float* y = (float*)d_out;                       // [N, H]

    char* ws = (char*)d_ws;
    size_t off = 0;
    auto carve = [&](size_t bytes) { void* p = ws + off; off = (off + bytes + 255) & ~(size_t)255; return p; };
    int*    tk_idx = (int*)   carve((size_t)Nn * Kk * 4);
    float*  tk_w   = (float*) carve((size_t)Nn * Kk * 4);
    int*    ecnt   = (int*)   carve((size_t)Ee * 4);
    int*    etok   = (int*)   carve((size_t)Ee * CAP * 4);
    float*  ewt    = (float*) carve((size_t)Ee * CAP * 4);
    __bf16* act_r  = (__bf16*)carve((size_t)Ee * CAP * Ii * 2);
    __bf16* act_s  = (__bf16*)carve((size_t)Nn * ISh * 2);
    __bf16* xb     = (__bf16*)carve((size_t)Nn * Hh * 2);
    __bf16* wgeb   = (__bf16*)carve((size_t)Ee * Ii * Hh * 2);
    __bf16* wueb   = (__bf16*)carve((size_t)Ee * Ii * Hh * 2);
    __bf16* wdeb   = (__bf16*)carve((size_t)Ee * Hh * Ii * 2);
    __bf16* wsgb   = (__bf16*)carve((size_t)ISh * Hh * 2);
    __bf16* wsub   = (__bf16*)carve((size_t)ISh * Hh * 2);
    __bf16* wsdb   = (__bf16*)carve((size_t)Hh * ISh * 2);

    // 0) fp32 -> bf16 mirrors (one pass; all later tiles are raw 16B copies)
    const int nE = Ee * Ii * Hh, nS = ISh * Hh, nX = Nn * Hh;
    moe_cvt_bf16<<<nX / 1024, 256, 0, stream>>>(x,       xb,   nX);
    moe_cvt_bf16<<<nE / 1024, 256, 0, stream>>>(we_gate, wgeb, nE);
    moe_cvt_bf16<<<nE / 1024, 256, 0, stream>>>(we_up,   wueb, nE);
    moe_cvt_bf16<<<nE / 1024, 256, 0, stream>>>(we_down, wdeb, nE);
    moe_cvt_bf16<<<nS / 1024, 256, 0, stream>>>(ws_gate, wsgb, nS);
    moe_cvt_bf16<<<nS / 1024, 256, 0, stream>>>(ws_up,   wsub, nS);
    moe_cvt_bf16<<<nS / 1024, 256, 0, stream>>>(ws_down, wsdb, nS);

    // 1) router: top-2 + normalized weights
    moe_router<<<Nn / 8, 256, 0, stream>>>(x, gate_w, tk_idx, tk_w);

    // 2) ordered capacity scan -> per-expert token/weight lists
    moe_scan<<<1, 256, 0, stream>>>(tk_idx, tk_w, etok, ewt, ecnt);

    // 3) shared expert up -> bf16 [N, ISh]
    moe_gemm_swiglu<false><<<dim3(ISh / 128, Nn / 32, 1), 256, 0, stream>>>(
        xb, wsgb, wsub, act_s, nullptr, nullptr, ISh, Hh, Nn);

    // 4) routed experts up (gathered) -> bf16 [E*CAP, I]
    moe_gemm_swiglu<true><<<dim3(Ii / 128, CAP / 32, Ee), 256, 0, stream>>>(
        xb, wgeb, wueb, act_r, etok, ecnt, Ii, Hh, 0);

    // 5) shared expert down: writes y (full overwrite)
    moe_gemm_down<false><<<dim3(Hh / 128, Nn / 32, 1), 256, 0, stream>>>(
        act_s, wsdb, y, nullptr, nullptr, nullptr, Hh, ISh, Nn);

    // 6) routed experts down: weighted atomic accumulate into y
    moe_gemm_down<true><<<dim3(Hh / 128, CAP / 32, Ee), 256, 0, stream>>>(
        act_r, wdeb, y, etok, ewt, ecnt, Hh, Ii, 0);
}
